// TransformerBlock_15470472200909
// MI455X (gfx1250) — matrix operand order repaired and checked by reading, not compiled, not run on hardware
//
#include <hip/hip_runtime.h>
#include <math.h>

typedef __bf16 bf16;
typedef __attribute__((ext_vector_type(16))) __bf16 v16bf;
typedef __attribute__((ext_vector_type(8)))  float  v8f;

#define HIDDEN 1024
#define QDIM   2048
#define NHQ    16
#define NHKV   8
#define HD     128
#define MLPD   3072
#define SEQ    2048
#define BATCH  2
#define ROWS   (BATCH*SEQ)   // 4096
#define EPSF   1e-6f

// ---------------- WMMA fragment helpers (ISA 7.12.2 layouts) ----------------

// A-matrix 16x32 bf16 tile from row-major [rows, ld], starting (row0, k0).
// Lane (m = lane&15, h = lane>>4): elems 0..7  = K k0+h*8+0..7
//                                  elems 8..15 = K k0+16+h*8+0..7
static __device__ __forceinline__ v16bf frag_a(const bf16* __restrict__ base,
                                               int ld, int row0, int k0) {
  int lane = threadIdx.x & 31;
  int hf = lane >> 4, m = lane & 15;
  const bf16* p = base + (size_t)(row0 + m) * ld + k0 + hf * 8;
  v16bf f;
#pragma unroll
  for (int i = 0; i < 8; ++i) { f[i] = p[i]; f[8 + i] = p[16 + i]; }
  return f;
}

// B-matrix 32x16 bf16 tile where column n is row (col0+n) of a row-major
// matrix W[ncols, ld] (i.e. computes A @ W^T).
// Lane (n = lane&15, h = lane>>4): elems 0..7  = K k0+h*8+0..7
//                                  elems 8..15 = K k0+16+h*8+0..7
static __device__ __forceinline__ v16bf frag_b(const bf16* __restrict__ base,
                                               int ld, int col0, int k0) {
  int lane = threadIdx.x & 31;
  int hf = lane >> 4, n = lane & 15;
  const bf16* p = base + (size_t)(col0 + n) * ld + k0 + hf * 8;
  v16bf f;
#pragma unroll
  for (int i = 0; i < 8; ++i) { f[i] = p[i]; f[8 + i] = p[16 + i]; }
  return f;
}

static __device__ __forceinline__ v8f wmma_bf16(v16bf a, v16bf b, v8f c) {
  return __builtin_amdgcn_wmma_f32_16x16x32_bf16(false, a, false, b,
                                                 (short)0, c, false, false);
}

// ---------------- elementwise / norm kernels ----------------

__global__ void cvt_f32_bf16(const float* __restrict__ s, bf16* __restrict__ d, int n) {
  int i = blockIdx.x * blockDim.x + threadIdx.x;
  int stride = gridDim.x * blockDim.x;
  for (; i < n; i += stride) d[i] = (bf16)s[i];
}

// One wave per row: RMSNorm(x, w) -> bf16
__global__ void rmsnorm_to_bf16(const float* __restrict__ X, const float* __restrict__ w,
                                bf16* __restrict__ Y, int ncols) {
  int warp = threadIdx.x >> 5, lane = threadIdx.x & 31;
  int row = blockIdx.x * (blockDim.x >> 5) + warp;
  const float* x = X + (size_t)row * ncols;
  float s = 0.f;
  for (int c = lane; c < ncols; c += 32) { float v = x[c]; s += v * v; }
#pragma unroll
  for (int off = 16; off > 0; off >>= 1) s += __shfl_xor(s, off, 32);
  float r = rsqrtf(s / (float)ncols + EPSF);
  bf16* y = Y + (size_t)row * ncols;
  for (int c = lane; c < ncols; c += 32) y[c] = (bf16)(x[c] * r * w[c]);
}

// One wave per (b,t,head): per-head RMSNorm + RoPE, write bf16 [B, nh, T, HD]
__global__ void qknorm_rope(const float* __restrict__ QK, const float* __restrict__ nw,
                            bf16* __restrict__ out, int nheads) {
  int wid = blockIdx.x * (blockDim.x >> 5) + (threadIdx.x >> 5);
  int lane = threadIdx.x & 31;
  int h = wid % nheads;
  int bt = wid / nheads;
  int b = bt / SEQ, t = bt % SEQ;
  const float* src = QK + (size_t)bt * ((size_t)nheads * HD) + (size_t)h * HD;
  float v0[2], v1[2];
  float s = 0.f;
#pragma unroll
  for (int i = 0; i < 2; ++i) {
    int j = lane * 2 + i;              // 0..63
    v0[i] = src[j]; v1[i] = src[j + 64];
    s += v0[i] * v0[i] + v1[i] * v1[i];
  }
#pragma unroll
  for (int off = 16; off > 0; off >>= 1) s += __shfl_xor(s, off, 32);
  float r = rsqrtf(s / (float)HD + EPSF);
  bf16* dst = out + (((size_t)b * nheads + h) * SEQ + t) * HD;
#pragma unroll
  for (int i = 0; i < 2; ++i) {
    int j = lane * 2 + i;
    float a  = v0[i] * r * nw[j];
    float bb = v1[i] * r * nw[j + 64];
    float ang = (float)t * __powf(10000.0f, -(float)j * (1.0f / 64.0f));
    float sn, cs;
    __sincosf(ang, &sn, &cs);
    dst[j]      = (bf16)(a * cs - bb * sn);
    dst[j + 64] = (bf16)(bb * cs + a * sn);
  }
}

// One wave per (b,t,kv-head): V fp32 [ROWS,1024] -> bf16 transposed [B,NHKV,HD,SEQ]
__global__ void v_to_bf16_t(const float* __restrict__ V, bf16* __restrict__ Vt) {
  int wid = blockIdx.x * (blockDim.x >> 5) + (threadIdx.x >> 5);
  int lane = threadIdx.x & 31;
  int h = wid % NHKV;
  int bt = wid / NHKV;
  int b = bt / SEQ, t = bt % SEQ;
  const float* src = V + (size_t)bt * HIDDEN + (size_t)h * HD;
  bf16* dst = Vt + (((size_t)b * NHKV + h) * HD) * SEQ + t;
#pragma unroll
  for (int i = 0; i < 4; ++i) {
    int d = lane * 4 + i;
    dst[(size_t)d * SEQ] = (bf16)src[d];
  }
}

// ---------------- GEMM kernels (C = A @ W^T) ----------------
// Register-blocked: each wave computes a 32x64 tile of C (8 WMMAs per k-step
// from 2 A-frags + 4 B-frags).  Block = 128 threads = 4 waves covering 64x128.
// grid = (N/128, M/64).

__global__ void gemm_bf16_f32(const bf16* __restrict__ A, const bf16* __restrict__ W,
                              float* __restrict__ C, int M, int N, int K) {
  int warp = threadIdx.x >> 5;
  int row0 = blockIdx.y * 64 + (warp >> 1) * 32;
  int col0 = blockIdx.x * 128 + (warp & 1) * 64;
  int lane = threadIdx.x & 31, hf = lane >> 4, n = lane & 15;
  v8f acc[2][4];
#pragma unroll
  for (int i = 0; i < 2; ++i)
#pragma unroll
    for (int j = 0; j < 4; ++j)
      acc[i][j] = (v8f){0.f, 0.f, 0.f, 0.f, 0.f, 0.f, 0.f, 0.f};
  for (int k = 0; k < K; k += 32) {
    if (k + 32 < K) {
      __builtin_prefetch(A + (size_t)(row0 + n) * K + k + 32, 0, 1);
      __builtin_prefetch(W + (size_t)(col0 + n) * K + k + 32, 0, 1);
    }
    v16bf a0 = frag_a(A, K, row0, k);
    v16bf a1 = frag_a(A, K, row0 + 16, k);
#pragma unroll
    for (int j = 0; j < 4; ++j) {
      v16bf b = frag_b(W, K, col0 + j * 16, k);
      acc[0][j] = wmma_bf16(a0, b, acc[0][j]);
      acc[1][j] = wmma_bf16(a1, b, acc[1][j]);
    }
  }
#pragma unroll
  for (int i = 0; i < 2; ++i)
#pragma unroll
    for (int j = 0; j < 4; ++j)
#pragma unroll
      for (int r = 0; r < 8; ++r)
        C[(size_t)(row0 + i * 16 + hf * 8 + r) * N + col0 + j * 16 + n] = acc[i][j][r];
}

__global__ void gemm_bf16_resid_f32(const bf16* __restrict__ A, const bf16* __restrict__ W,
                                    const float* __restrict__ resid, float* __restrict__ C,
                                    int M, int N, int K) {
  int warp = threadIdx.x >> 5;
  int row0 = blockIdx.y * 64 + (warp >> 1) * 32;
  int col0 = blockIdx.x * 128 + (warp & 1) * 64;
  int lane = threadIdx.x & 31, hf = lane >> 4, n = lane & 15;
  v8f acc[2][4];
#pragma unroll
  for (int i = 0; i < 2; ++i)
#pragma unroll
    for (int j = 0; j < 4; ++j)
      acc[i][j] = (v8f){0.f, 0.f, 0.f, 0.f, 0.f, 0.f, 0.f, 0.f};
  for (int k = 0; k < K; k += 32) {
    if (k + 32 < K) {
      __builtin_prefetch(A + (size_t)(row0 + n) * K + k + 32, 0, 1);
      __builtin_prefetch(W + (size_t)(col0 + n) * K + k + 32, 0, 1);
    }
    v16bf a0 = frag_a(A, K, row0, k);
    v16bf a1 = frag_a(A, K, row0 + 16, k);
#pragma unroll
    for (int j = 0; j < 4; ++j) {
      v16bf b = frag_b(W, K, col0 + j * 16, k);
      acc[0][j] = wmma_bf16(a0, b, acc[0][j]);
      acc[1][j] = wmma_bf16(a1, b, acc[1][j]);
    }
  }
#pragma unroll
  for (int i = 0; i < 2; ++i)
#pragma unroll
    for (int j = 0; j < 4; ++j)
#pragma unroll
      for (int r = 0; r < 8; ++r) {
        size_t idx = (size_t)(row0 + i * 16 + hf * 8 + r) * N + col0 + j * 16 + n;
        C[idx] = acc[i][j][r] + resid[idx];
      }
}

// Fused gate/up GEMM: H = silu(A@Wg^T) * (A@Wu^T), bf16 out.
// Each wave: 16x64 tile of BOTH gemms (8 WMMAs per k-step from 1 A-frag).
// grid = (N/128, M/32), block = 128.
__global__ void gemm_gateup(const bf16* __restrict__ A, const bf16* __restrict__ Wg,
                            const bf16* __restrict__ Wu, bf16* __restrict__ H,
                            int M, int N, int K) {
  int warp = threadIdx.x >> 5;
  int row0 = blockIdx.y * 32 + (warp >> 1) * 16;
  int col0 = blockIdx.x * 128 + (warp & 1) * 64;
  int lane = threadIdx.x & 31, hf = lane >> 4, n = lane & 15;
  v8f accg[4], accu[4];
#pragma unroll
  for (int j = 0; j < 4; ++j) {
    accg[j] = (v8f){0.f, 0.f, 0.f, 0.f, 0.f, 0.f, 0.f, 0.f};
    accu[j] = (v8f){0.f, 0.f, 0.f, 0.f, 0.f, 0.f, 0.f, 0.f};
  }
  for (int k = 0; k < K; k += 32) {
    if (k + 32 < K) {
      __builtin_prefetch(A + (size_t)(row0 + n) * K + k + 32, 0, 1);
      __builtin_prefetch(Wg + (size_t)(col0 + n) * K + k + 32, 0, 1);
      __builtin_prefetch(Wu + (size_t)(col0 + n) * K + k + 32, 0, 1);
    }
    v16bf a = frag_a(A, K, row0, k);
#pragma unroll
    for (int j = 0; j < 4; ++j) {
      accg[j] = wmma_bf16(a, frag_b(Wg, K, col0 + j * 16, k), accg[j]);
      accu[j] = wmma_bf16(a, frag_b(Wu, K, col0 + j * 16, k), accu[j]);
    }
  }
#pragma unroll
  for (int j = 0; j < 4; ++j)
#pragma unroll
    for (int r = 0; r < 8; ++r) {
      float g = accg[j][r];
      float sg = g / (1.0f + __expf(-g));     // silu
      H[(size_t)(row0 + hf * 8 + r) * N + col0 + j * 16 + n] = (bf16)(sg * accu[j][r]);
    }
}

// ---------------- Flash attention (causal, GQA) ----------------
// One wave per (b, h, 16-query tile). Computes S^T = K·Q^T so the softmaxed
// P values are already in A-fragment layout for P·V (no cross-lane transpose).
__global__ void __launch_bounds__(32)
flash_attn(const bf16* __restrict__ Qb,   // [B, NHQ,  SEQ, HD]
           const bf16* __restrict__ Kb,   // [B, NHKV, SEQ, HD]
           const bf16* __restrict__ Vt,   // [B, NHKV, HD,  SEQ]
           bf16* __restrict__ O) {        // [B*SEQ, QDIM]
  int wid = blockIdx.x;
  int ntile = SEQ / 16;
  int qt = wid % ntile;
  int tmp = wid / ntile;
  int h = tmp % NHQ;
  int b = tmp / NHQ;
  int kvh = h >> 1;                        // groups = NHQ/NHKV = 2
  int i0 = qt * 16;
  int lane = threadIdx.x & 31, hf = lane >> 4, nn = lane & 15;

  const bf16* Qh = Qb + (((size_t)b * NHQ  + h)   * SEQ) * HD;
  const bf16* Kh = Kb + (((size_t)b * NHKV + kvh) * SEQ) * HD;
  const bf16* Vh = Vt + (((size_t)b * NHKV + kvh) * HD)  * SEQ;

  v16bf qfrag[4];
#pragma unroll
  for (int d = 0; d < 4; ++d) qfrag[d] = frag_b(Qh, HD, i0, d * 32);

  v8f o[8];
#pragma unroll
  for (int c = 0; c < 8; ++c) o[c] = (v8f){0.f, 0.f, 0.f, 0.f, 0.f, 0.f, 0.f, 0.f};

  float mstate = -3.0e38f, lstate = 0.f;   // per-lane query = i0 + nn
  const float scale = 0.08838834764831845f; // 1/sqrt(128)
  int q_of_lane = i0 + nn;
  int base8 = hf * 8;

  for (int kb = 0; kb < i0 + 16; kb += 32) {
    // S^T tiles: st0 rows = keys kb..kb+15, st1 rows = keys kb+16..kb+31,
    // columns = queries i0..i0+15
    v8f st0 = {0.f, 0.f, 0.f, 0.f, 0.f, 0.f, 0.f, 0.f};
    v8f st1 = {0.f, 0.f, 0.f, 0.f, 0.f, 0.f, 0.f, 0.f};
#pragma unroll
    for (int d = 0; d < 4; ++d) {
      st0 = wmma_bf16(frag_a(Kh, HD, kb,      d * 32), qfrag[d], st0);
      st1 = wmma_bf16(frag_a(Kh, HD, kb + 16, d * 32), qfrag[d], st1);
    }
    // scale + causal mask; per-lane holds 16 of the 32 key scores for query nn
    float p0[8], p1[8];
    float lm = -3.0e38f;
#pragma unroll
    for (int r = 0; r < 8; ++r) {
      int key0 = kb + hf * 8 + r;
      int key1 = key0 + 16;
      p0[r] = (key0 <= q_of_lane) ? st0[r] * scale : -3.0e38f;
      p1[r] = (key1 <= q_of_lane) ? st1[r] * scale : -3.0e38f;
      lm = fmaxf(lm, fmaxf(p0[r], p1[r]));
    }
    lm = fmaxf(lm, __shfl_xor(lm, 16, 32));      // merge key-halves
    float newm = fmaxf(mstate, lm);
    float corr = __expf(mstate - newm);
    float ls = 0.f;
    v16bf pa;                                    // A-fragment of P (16q x 32k)
#pragma unroll
    for (int r = 0; r < 8; ++r) {
      float e0 = __expf(p0[r] - newm);
      float e1 = __expf(p1[r] - newm);
      ls += e0 + e1;
      pa[r]     = (bf16)e0;
      pa[8 + r] = (bf16)e1;
    }
    ls += __shfl_xor(ls, 16, 32);
    lstate = lstate * corr + ls;
    mstate = newm;

    // broadcast per-query rescale factor to output-accumulator row layout
    float corr_row[8];
#pragma unroll
    for (int r = 0; r < 8; ++r) corr_row[r] = __shfl(corr, base8 + r, 32);
#pragma unroll
    for (int c = 0; c < 8; ++c) {
#pragma unroll
      for (int r = 0; r < 8; ++r) o[c][r] *= corr_row[r];
      o[c] = wmma_bf16(pa, frag_b(Vh, SEQ, c * 16, kb), o[c]);
    }
  }

  float linv[8];
#pragma unroll
  for (int r = 0; r < 8; ++r) linv[r] = 1.0f / __shfl(lstate, base8 + r, 32);
#pragma unroll
  for (int c = 0; c < 8; ++c) {
#pragma unroll
    for (int r = 0; r < 8; ++r) {
      int t = i0 + hf * 8 + r;
      O[((size_t)b * SEQ + t) * QDIM + (size_t)h * HD + c * 16 + nn] =
          (bf16)(o[c][r] * linv[r]);
    }
  }
}

// ---------------- host orchestration ----------------

extern "C" void kernel_launch(void* const* d_in, const int* in_sizes, int n_in,
                              void* d_out, int out_size, void* d_ws, size_t ws_size,
                              hipStream_t stream) {
  (void)in_sizes; (void)n_in; (void)out_size; (void)ws_size;
  const float* x   = (const float*)d_in[0];
  const float* wq  = (const float*)d_in[1];
  const float* wk  = (const float*)d_in[2];
  const float* wv  = (const float*)d_in[3];
  const float* wo  = (const float*)d_in[4];
  const float* qn  = (const float*)d_in[5];
  const float* kn  = (const float*)d_in[6];
  const float* ln1 = (const float*)d_in[7];
  const float* ln2 = (const float*)d_in[8];
  const float* wg  = (const float*)d_in[9];
  const float* wu  = (const float*)d_in[10];
  const float* wd  = (const float*)d_in[11];
  float* out = (float*)d_out;

  char* ws = (char*)d_ws;
  size_t off = 0;
  auto alloc = [&](size_t bytes) -> void* {
    off = (off + 255) & ~(size_t)255;
    void* p = ws + off;
    off += bytes;
    return p;
  };

  bf16* wq_bf  = (bf16*)alloc((size_t)QDIM * HIDDEN * 2);
  bf16* wk_bf  = (bf16*)alloc((size_t)HIDDEN * HIDDEN * 2);
  bf16* wv_bf  = (bf16*)alloc((size_t)HIDDEN * HIDDEN * 2);
  bf16* wo_bf  = (bf16*)alloc((size_t)HIDDEN * QDIM * 2);
  bf16* wg_bf  = (bf16*)alloc((size_t)MLPD * HIDDEN * 2);
  bf16* wu_bf  = (bf16*)alloc((size_t)MLPD * HIDDEN * 2);
  bf16* wd_bf  = (bf16*)alloc((size_t)HIDDEN * MLPD * 2);
  bf16* xn_bf  = (bf16*)alloc((size_t)ROWS * HIDDEN * 2);
  float* qf    = (float*)alloc((size_t)ROWS * QDIM * 4);
  float* kf    = (float*)alloc((size_t)ROWS * HIDDEN * 4);
  float* vf    = (float*)alloc((size_t)ROWS * HIDDEN * 4);
  bf16* qbf    = (bf16*)alloc((size_t)BATCH * NHQ  * SEQ * HD * 2);
  bf16* kbf    = (bf16*)alloc((size_t)BATCH * NHKV * SEQ * HD * 2);
  bf16* vtbf   = (bf16*)alloc((size_t)BATCH * NHKV * HD * SEQ * 2);
  bf16* attnbf = (bf16*)alloc((size_t)ROWS * QDIM * 2);
  float* xattn = (float*)alloc((size_t)ROWS * HIDDEN * 4);
  bf16* hbf    = (bf16*)alloc((size_t)ROWS * HIDDEN * 2);
  bf16* hidbf  = (bf16*)alloc((size_t)ROWS * MLPD * 2);

  // 0) weights -> bf16
  cvt_f32_bf16<<<1024, 256, 0, stream>>>(wq, wq_bf, QDIM * HIDDEN);
  cvt_f32_bf16<<<1024, 256, 0, stream>>>(wk, wk_bf, HIDDEN * HIDDEN);
  cvt_f32_bf16<<<1024, 256, 0, stream>>>(wv, wv_bf, HIDDEN * HIDDEN);
  cvt_f32_bf16<<<1024, 256, 0, stream>>>(wo, wo_bf, HIDDEN * QDIM);
  cvt_f32_bf16<<<1024, 256, 0, stream>>>(wg, wg_bf, MLPD * HIDDEN);
  cvt_f32_bf16<<<1024, 256, 0, stream>>>(wu, wu_bf, MLPD * HIDDEN);
  cvt_f32_bf16<<<1024, 256, 0, stream>>>(wd, wd_bf, HIDDEN * MLPD);

  // 1) x_norm = rmsnorm(x, ln1) -> bf16
  rmsnorm_to_bf16<<<ROWS / 8, 256, 0, stream>>>(x, ln1, xn_bf, HIDDEN);

  // 2) QKV projections (bf16 WMMA, fp32 out)
  gemm_bf16_f32<<<dim3(QDIM / 128, ROWS / 64), 128, 0, stream>>>(
      xn_bf, wq_bf, qf, ROWS, QDIM, HIDDEN);
  gemm_bf16_f32<<<dim3(HIDDEN / 128, ROWS / 64), 128, 0, stream>>>(
      xn_bf, wk_bf, kf, ROWS, HIDDEN, HIDDEN);
  gemm_bf16_f32<<<dim3(HIDDEN / 128, ROWS / 64), 128, 0, stream>>>(
      xn_bf, wv_bf, vf, ROWS, HIDDEN, HIDDEN);

  // 3) QK-norm + RoPE + head-major bf16; V transpose
  qknorm_rope<<<(ROWS * NHQ) / 8, 256, 0, stream>>>(qf, qn, qbf, NHQ);
  qknorm_rope<<<(ROWS * NHKV) / 8, 256, 0, stream>>>(kf, kn, kbf, NHKV);
  v_to_bf16_t<<<(ROWS * NHKV) / 8, 256, 0, stream>>>(vf, vtbf);

  // 4) causal flash attention
  flash_attn<<<BATCH * NHQ * (SEQ / 16), 32, 0, stream>>>(qbf, kbf, vtbf, attnbf);

  // 5) x_attn = x + attn @ wo^T
  gemm_bf16_resid_f32<<<dim3(HIDDEN / 128, ROWS / 64), 128, 0, stream>>>(
      attnbf, wo_bf, x, xattn, ROWS, HIDDEN, QDIM);

  // 6) h = rmsnorm(x_attn, ln2)
  rmsnorm_to_bf16<<<ROWS / 8, 256, 0, stream>>>(xattn, ln2, hbf, HIDDEN);

  // 7) hidden = silu(h@wg^T) * (h@wu^T)
  gemm_gateup<<<dim3(MLPD / 128, ROWS / 32), 128, 0, stream>>>(
      hbf, wg_bf, wu_bf, hidbf, ROWS, MLPD, HIDDEN);

  // 8) out = x_attn + hidden @ wd^T
  gemm_bf16_resid_f32<<<dim3(HIDDEN / 128, ROWS / 64), 128, 0, stream>>>(
      hidbf, wd_bf, xattn, out, ROWS, HIDDEN, MLPD);
}
